// PIAA_ICI_9079560863794
// MI455X (gfx1250) — matrix operand order repaired and checked by reading, not compiled, not run on hardware
//
#include <hip/hip_runtime.h>
#include <hip/hip_bf16.h>
#include <math.h>

#define B_SZ 2048
#define FEAT 2048
#define NUM_BINS 9
#define NUM_ATTR 12
#define NUM_PT 20
#define DD 64
#define HH 256

typedef __attribute__((ext_vector_type(16))) _Float16 v16h;
typedef __attribute__((ext_vector_type(8)))  _Float16 v8h;
typedef __attribute__((ext_vector_type(8)))  float    v8f;

static __device__ __forceinline__ v16h combine8(v8h lo, v8h hi) {
    return __builtin_shufflevector(lo, hi, 0, 1, 2, 3, 4, 5, 6, 7,
                                           8, 9, 10, 11, 12, 13, 14, 15);
}
static __device__ __forceinline__ v8f wmma16(v16h a, v16h b, v8f c) {
    return __builtin_amdgcn_wmma_f32_16x16x32_f16(false, a, false, b,
                                                  (short)0, c, false, false);
}

// ---------------------------------------------------------------------------
// Pack a [K,N] f32 weight matrix into WMMA B-fragment order (f16):
// packed[((nt*(K/32)+ks)*32 + lane)*16 + e] = W[(ks*32+(lane>>4)*8+(e&7)+(e>>3)*16)*N + nt*16+(lane&15)]
// so each lane's 16 B-values are one contiguous 32-byte run.
// ---------------------------------------------------------------------------
__global__ void pack_wmma_b(const float* __restrict__ W, _Float16* __restrict__ out,
                            int K, int N) {
    int idx = blockIdx.x * 256 + threadIdx.x;
    if (idx >= K * N) return;
    int kk   = idx & 15;
    int lane = (idx >> 4) & 31;
    int tile = idx >> 9;
    int nkt  = K >> 5;
    int nt   = tile / nkt, ks = tile - nt * nkt;
    int k = ks * 32 + ((lane >> 4) << 3) + (kk & 7) + ((kk >> 3) << 4);
    int n = nt * 16 + (lane & 15);
    out[idx] = (_Float16)W[k * N + n];
}

// ---------------------------------------------------------------------------
// Kernel A: per image row -> logit(9), attr_raw(12), softmax*scale, direct MLP,
//           and the attr_img node MLP (12->256 relu ->768). One block per row.
// ---------------------------------------------------------------------------
__global__ void nima_direct_attrimg_kernel(
    const float* __restrict__ images,
    const float* __restrict__ nima_Wl, const float* __restrict__ nima_bl,
    const float* __restrict__ nima_Wa, const float* __restrict__ nima_ba,
    const float* __restrict__ dist_W1, const float* __restrict__ dist_b1,
    const float* __restrict__ dist_W2, const float* __restrict__ dist_b2,
    const float* __restrict__ img_W1, const float* __restrict__ img_b1,
    const float* __restrict__ img_W2, const float* __restrict__ img_b2,
    float* __restrict__ attr_img, float* __restrict__ direct)
{
    __shared__ float row[FEAT];
    __shared__ float red[21][256];
    __shared__ float logitArr[NUM_BINS];
    __shared__ float smArr[NUM_BINS];
    __shared__ float araw[NUM_ATTR];
    __shared__ float hid[HH];

    const int b = blockIdx.x;
    const int t = threadIdx.x;

    for (int i = t; i < FEAT; i += 256) row[i] = images[b * FEAT + i];
    __syncthreads();

    float acc[21];
#pragma unroll
    for (int c = 0; c < 21; ++c) acc[c] = 0.f;
    const int k0 = t * 8;
#pragma unroll
    for (int k = 0; k < 8; ++k) {
        float x = row[k0 + k];
#pragma unroll
        for (int c = 0; c < NUM_BINS; ++c) acc[c]     += x * nima_Wl[(k0 + k) * NUM_BINS + c];
#pragma unroll
        for (int c = 0; c < NUM_ATTR; ++c) acc[9 + c] += x * nima_Wa[(k0 + k) * NUM_ATTR + c];
    }
#pragma unroll
    for (int c = 0; c < 21; ++c) red[c][t] = acc[c];
    __syncthreads();
    for (int s = 128; s > 0; s >>= 1) {
        if (t < s) {
#pragma unroll
            for (int c = 0; c < 21; ++c) red[c][t] += red[c][t + s];
        }
        __syncthreads();
    }
    if (t < NUM_BINS) logitArr[t] = red[t][0] + nima_bl[t];
    if (t < NUM_ATTR) araw[t] = red[9 + t][0] + nima_ba[t];
    __syncthreads();

    if (t == 0) {
        float m = logitArr[0];
        for (int c = 1; c < NUM_BINS; ++c) m = fmaxf(m, logitArr[c]);
        float s = 0.f, e[NUM_BINS];
        for (int c = 0; c < NUM_BINS; ++c) { e[c] = expf(logitArr[c] - m); s += e[c]; }
        for (int c = 0; c < NUM_BINS; ++c) smArr[c] = (e[c] / s) * (1.0f + 0.5f * (float)c);
    }
    __syncthreads();

    {   // direct head
        float hd = dist_b1[t];
#pragma unroll
        for (int c = 0; c < NUM_BINS; ++c) hd += smArr[c] * dist_W1[c * HH + t];
        hd = hd > 0.f ? hd : 0.f;
        red[0][t] = hd * dist_W2[t];
        __syncthreads();
        for (int s = 128; s > 0; s >>= 1) {
            if (t < s) red[0][t] += red[0][t + s];
            __syncthreads();
        }
        if (t == 0) direct[b] = red[0][0] + dist_b2[0];
    }

    {   // attr_img node MLP
        float hv = img_b1[t];
#pragma unroll
        for (int c = 0; c < NUM_ATTR; ++c) hv += araw[c] * img_W1[c * HH + t];
        hid[t] = hv > 0.f ? hv : 0.f;
        __syncthreads();
        for (int oi = t; oi < NUM_ATTR * DD; oi += 256) {
            float o = img_b2[oi];
            for (int k = 0; k < HH; ++k) o += hid[k] * img_W2[k * (NUM_ATTR * DD) + oi];
            attr_img[b * (NUM_ATTR * DD) + oi] = o;
        }
    }
}

// ---------------------------------------------------------------------------
// Kernel B: attr_user node MLP
// ---------------------------------------------------------------------------
__global__ void attr_user_kernel(
    const float* __restrict__ traits,
    const float* __restrict__ W1, const float* __restrict__ b1,
    const float* __restrict__ W2, const float* __restrict__ b2,
    float* __restrict__ attr_user)
{
    __shared__ float pt[NUM_PT];
    __shared__ float hid[HH];
    const int b = blockIdx.x;
    const int t = threadIdx.x;
    if (t < NUM_PT) pt[t] = traits[b * NUM_PT + t];
    __syncthreads();
    float hv = b1[t];
#pragma unroll
    for (int c = 0; c < NUM_PT; ++c) hv += pt[c] * W1[c * HH + t];
    hid[t] = hv > 0.f ? hv : 0.f;
    __syncthreads();
    for (int oi = t; oi < NUM_ATTR * DD; oi += 256) {
        float o = b2[oi];
        for (int k = 0; k < HH; ++k) o += hid[k] * W2[k * (NUM_ATTR * DD) + oi];
        attr_user[b * (NUM_ATTR * DD) + oi] = o;
    }
}

// ---------------------------------------------------------------------------
// Kernel C: per-batch attribute sums
// ---------------------------------------------------------------------------
__global__ void attr_sums_kernel(const float* __restrict__ attr_img,
                                 const float* __restrict__ attr_user,
                                 float* __restrict__ img_sum,
                                 float* __restrict__ user_sum)
{
    const int b = blockIdx.x;
    const int t = threadIdx.x;  // 128
    const float* src = (t < 64) ? attr_img : attr_user;
    float* dst = (t < 64) ? img_sum : user_sum;
    const int d = t & 63;
    float s = 0.f;
#pragma unroll
    for (int j = 0; j < NUM_ATTR; ++j) s += src[b * 768 + j * 64 + d];
    dst[b * 64 + d] = s;
}

// ---------------------------------------------------------------------------
// Kernel D (WMMA): internal interaction, one workgroup (8 waves) per batch.
// A-fragments: 2 x 16B contiguous LDS chunks; B-fragments: pre-packed, 32B run.
// ---------------------------------------------------------------------------
__global__ void internal_interaction_kernel(
    const float* __restrict__ attr,      // [B,768]
    const _Float16* __restrict__ W1p,    // packed (K=64,N=256)
    const float* __restrict__ b1,
    const _Float16* __restrict__ W2p,    // packed (K=256,N=64)
    const float* __restrict__ b2,
    float* __restrict__ outp)            // [B,768]
{
    extern __shared__ char smem[];
    _Float16* P  = (_Float16*)smem;                  // [144][72] f16 (GEMM1 A)
    float*    O  = (float*)smem;                     // [144][64] f32 (overlay)
    _Float16* W1 = (_Float16*)(smem + 36864);        // packed, 32768 B
    _Float16* W2 = (_Float16*)(smem + 69632);        // packed, 32768 B
    _Float16* H  = (_Float16*)(smem + 102400);       // [144][264] f16, 76032 B
    float* b1s   = (float*)(smem + 178432);          // 256
    float* b2s   = (float*)(smem + 179456);          // 64
    float* e     = (float*)(smem + 179712);          // 768  -> total 182784 B

    const int t = threadIdx.x;
    const int b = blockIdx.x;
    const int lane = t & 31;
    const int wave = t >> 5;
    const int colin = lane & 15;
    const int khalf = (lane >> 4) * 8;
    const int rb    = (lane >> 4) * 8;

    for (int i = t; i < 768; i += 256) e[i] = attr[b * 768 + i];
    {
        const uint4* s1 = (const uint4*)W1p; uint4* d1 = (uint4*)W1;
        for (int i = t; i < 32768 / 16; i += 256) d1[i] = s1[i];
        const uint4* s2 = (const uint4*)W2p; uint4* d2 = (uint4*)W2;
        for (int i = t; i < 32768 / 16; i += 256) d2[i] = s2[i];
    }
    b1s[t] = b1[t];
    if (t < 64) b2s[t] = b2[t];
    __syncthreads();

    // pair matrix P[144][72] (f16)
    for (int idx = t; idx < 144 * 64; idx += 256) {
        int m = idx >> 6, d = idx & 63;
        int i = m / 12, j = m - i * 12;
        P[m * 72 + d] = (_Float16)(e[i * 64 + d] * e[j * 64 + d]);
    }
    __syncthreads();

    // ---- GEMM1: H = relu(P @ W1 + b1); M=144 (9 tiles), N=256, K=64.
    // Register blocked: job = (mt, group of 4 nt); A frags reused 4x.
    for (int job = wave; job < 36; job += 8) {
        int mt = job >> 2, ng = job & 3;
        v16h a[2];
#pragma unroll
        for (int ks = 0; ks < 2; ++ks) {
            const _Float16* pa = P + (mt * 16 + colin) * 72 + ks * 32 + khalf;
            a[ks] = combine8(*(const v8h*)pa, *(const v8h*)(pa + 16));
        }
        v8f acc[4];
#pragma unroll
        for (int q = 0; q < 4; ++q) acc[q] = (v8f){};
#pragma unroll
        for (int q = 0; q < 4; ++q) {
            int nt = ng * 4 + q;
#pragma unroll
            for (int ks = 0; ks < 2; ++ks) {
                const _Float16* pb = W1 + ((nt * 2 + ks) * 32 + lane) * 16;
                acc[q] = wmma16(a[ks], combine8(*(const v8h*)pb, *(const v8h*)(pb + 8)), acc[q]);
            }
        }
#pragma unroll
        for (int q = 0; q < 4; ++q) {
            int col = (ng * 4 + q) * 16 + colin;
            float bias = b1s[col];
#pragma unroll
            for (int r = 0; r < 8; ++r) {
                float hv = acc[q][r] + bias;
                H[(mt * 16 + rb + r) * 264 + col] = (_Float16)(hv > 0.f ? hv : 0.f);
            }
        }
    }
    __syncthreads();

    // ---- GEMM2: O = H @ W2 + b2; M=144, N=64 (4 tiles), K=256 (8 steps).
    for (int job = wave; job < 9; job += 8) {
        int mt = job;
        v8f acc[4];
#pragma unroll
        for (int q = 0; q < 4; ++q) acc[q] = (v8f){};
        for (int ks = 0; ks < 8; ++ks) {
            const _Float16* pa = H + (mt * 16 + colin) * 264 + ks * 32 + khalf;
            v16h a = combine8(*(const v8h*)pa, *(const v8h*)(pa + 16));
#pragma unroll
            for (int q = 0; q < 4; ++q) {
                const _Float16* pb = W2 + ((q * 8 + ks) * 32 + lane) * 16;
                acc[q] = wmma16(a, combine8(*(const v8h*)pb, *(const v8h*)(pb + 8)), acc[q]);
            }
        }
#pragma unroll
        for (int q = 0; q < 4; ++q) {
            int col = q * 16 + colin;
            float bias = b2s[col];
#pragma unroll
            for (int r = 0; r < 8; ++r)
                O[(mt * 16 + rb + r) * 64 + col] = acc[q][r] + bias;
        }
    }
    __syncthreads();

    // ---- reduce over i
    for (int idx = t; idx < 768; idx += 256) {
        int j = idx >> 6, d = idx & 63;
        float s = 0.f;
#pragma unroll
        for (int i = 0; i < NUM_ATTR; ++i) s += O[(i * 12 + j) * 64 + d];
        outp[b * 768 + idx] = s;
    }
}

// ---------------------------------------------------------------------------
// Kernel E (WMMA): 3-step GRU. Block = 8 waves, 128 rows; each wave owns one
// 16-row M-tile. Gates computed as 16x16x32 WMMAs; h lives in C-fragment regs.
// ---------------------------------------------------------------------------
__global__ void gru_wmma_kernel(
    const float* __restrict__ attr,      // init [B*12,64]
    const float* __restrict__ internal_, // [B*12,64]
    const float* __restrict__ osum,      // [B,64]
    const _Float16* __restrict__ Wih_p,  // packed (K=64,N=192): 24 tiles*512
    const _Float16* __restrict__ Whh_p,
    const float* __restrict__ bih, const float* __restrict__ bhh,
    float* __restrict__ fused)           // [B*12,64]
{
    extern __shared__ char smem[];
    _Float16* wih = (_Float16*)smem;              // 24576 B
    _Float16* whh = (_Float16*)(smem + 24576);    // 24576 B
    _Float16* xl  = (_Float16*)(smem + 49152);    // [128][72] 18432 B
    _Float16* hl  = (_Float16*)(smem + 67584);    // [128][72] 18432 B
    float* bihs   = (float*)(smem + 86016);       // 192
    float* bhhs   = (float*)(smem + 86784);       // 192  -> total 87552 B

    const int t = threadIdx.x;
    const int lane = t & 31;
    const int wave = t >> 5;
    const int rowBase = blockIdx.x * 128;
    const int colin = lane & 15;
    const int khalf = (lane >> 4) * 8;
    const int rb    = (lane >> 4) * 8;
    const int wrow  = wave * 16;

    {
        const uint4* s1 = (const uint4*)Wih_p; uint4* d1 = (uint4*)wih;
        for (int i = t; i < 24576 / 16; i += 256) d1[i] = s1[i];
        const uint4* s2 = (const uint4*)Whh_p; uint4* d2 = (uint4*)whh;
        for (int i = t; i < 24576 / 16; i += 256) d2[i] = s2[i];
    }
    if (t < 192) { bihs[t] = bih[t]; bhhs[t] = bhh[t]; }

    v8f hfrag[4];
#pragma unroll
    for (int q = 0; q < 4; ++q) hfrag[q] = (v8f){};

    for (int step = 0; step < 3; ++step) {
        __syncthreads();   // previous step's readers done before xl rewrite
        for (int idx = t; idx < 128 * 64; idx += 256) {
            int r = idx >> 6, d = idx & 63;
            int grow = rowBase + r;
            float v;
            if (step == 0)      v = attr[grow * 64 + d];
            else if (step == 1) v = internal_[grow * 64 + d];
            else                v = attr[grow * 64 + d] * osum[(grow / 12) * 64 + d];
            xl[r * 72 + d] = (_Float16)v;
        }
        __syncthreads();

        v16h ax[2], ah[2];
#pragma unroll
        for (int ks = 0; ks < 2; ++ks) {
            const _Float16* px = xl + (wrow + colin) * 72 + ks * 32 + khalf;
            ax[ks] = combine8(*(const v8h*)px, *(const v8h*)(px + 16));
            if (step > 0) {
                const _Float16* ph = hl + (wrow + colin) * 72 + ks * 32 + khalf;
                ah[ks] = combine8(*(const v8h*)ph, *(const v8h*)(ph + 16));
            }
        }

#pragma unroll
        for (int q = 0; q < 4; ++q) {
            v8f gi0 = {}, gi4 = {}, gi8 = {}, gh0 = {}, gh4 = {}, gh8 = {};
#pragma unroll
            for (int ks = 0; ks < 2; ++ks) {
                const _Float16* p0 = wih + (((q    ) * 2 + ks) * 32 + lane) * 16;
                const _Float16* p4 = wih + (((q + 4) * 2 + ks) * 32 + lane) * 16;
                const _Float16* p8 = wih + (((q + 8) * 2 + ks) * 32 + lane) * 16;
                gi0 = wmma16(ax[ks], combine8(*(const v8h*)p0, *(const v8h*)(p0 + 8)), gi0);
                gi4 = wmma16(ax[ks], combine8(*(const v8h*)p4, *(const v8h*)(p4 + 8)), gi4);
                gi8 = wmma16(ax[ks], combine8(*(const v8h*)p8, *(const v8h*)(p8 + 8)), gi8);
                if (step > 0) {
                    const _Float16* q0 = whh + (((q    ) * 2 + ks) * 32 + lane) * 16;
                    const _Float16* q4 = whh + (((q + 4) * 2 + ks) * 32 + lane) * 16;
                    const _Float16* q8 = whh + (((q + 8) * 2 + ks) * 32 + lane) * 16;
                    gh0 = wmma16(ah[ks], combine8(*(const v8h*)q0, *(const v8h*)(q0 + 8)), gh0);
                    gh4 = wmma16(ah[ks], combine8(*(const v8h*)q4, *(const v8h*)(q4 + 8)), gh4);
                    gh8 = wmma16(ah[ks], combine8(*(const v8h*)q8, *(const v8h*)(q8 + 8)), gh8);
                }
            }
            const int c0 = q * 16 + colin;
            const float bi0 = bihs[c0], bi4 = bihs[64 + c0], bi8 = bihs[128 + c0];
            const float bh0 = bhhs[c0], bh4 = bhhs[64 + c0], bh8 = bhhs[128 + c0];
#pragma unroll
            for (int r = 0; r < 8; ++r) {
                float ir = gi0[r] + bi0, hr = gh0[r] + bh0;
                float iz = gi4[r] + bi4, hz = gh4[r] + bh4;
                float in_ = gi8[r] + bi8, hn = gh8[r] + bh8;
                float rg = 1.f / (1.f + expf(-(ir + hr)));
                float zg = 1.f / (1.f + expf(-(iz + hz)));
                float ng = tanhf(in_ + rg * hn);
                hfrag[q][r] = (1.f - zg) * ng + zg * hfrag[q][r];
            }
            if (step < 2) {
#pragma unroll
                for (int r = 0; r < 8; ++r)
                    hl[(wrow + rb + r) * 72 + c0] = (_Float16)hfrag[q][r];
            }
        }
    }

#pragma unroll
    for (int q = 0; q < 4; ++q) {
        int d = q * 16 + colin;
#pragma unroll
        for (int r = 0; r < 8; ++r)
            fused[(rowBase + wrow + rb + r) * 64 + d] = hfrag[q][r];
    }
}

// ---------------------------------------------------------------------------
// Kernel F: final reduce
// ---------------------------------------------------------------------------
__global__ void final_kernel(const float* __restrict__ fused_img,
                             const float* __restrict__ fused_user,
                             const float* __restrict__ corr_W,
                             const float* __restrict__ corr_b,
                             const float* __restrict__ direct,
                             float* __restrict__ out)
{
    __shared__ float red[64];
    const int b = blockIdx.x, t = threadIdx.x;
    float s = 0.f;
#pragma unroll
    for (int j = 0; j < NUM_ATTR; ++j)
        s += fused_img[b * 768 + j * 64 + t] + fused_user[b * 768 + j * 64 + t];
    red[t] = s * corr_W[t];
    __syncthreads();
    for (int st = 32; st > 0; st >>= 1) {
        if (t < st) red[t] += red[t + st];
        __syncthreads();
    }
    if (t == 0) out[b] = red[0] + corr_b[0] + direct[b];
}

// ---------------------------------------------------------------------------
extern "C" void kernel_launch(void* const* d_in, const int* in_sizes, int n_in,
                              void* d_out, int out_size, void* d_ws, size_t ws_size,
                              hipStream_t stream) {
    const float* images   = (const float*)d_in[0];
    const float* traits   = (const float*)d_in[1];
    const float* nima_Wl  = (const float*)d_in[2];
    const float* nima_bl  = (const float*)d_in[3];
    const float* nima_Wa  = (const float*)d_in[4];
    const float* nima_ba  = (const float*)d_in[5];
    const float* img_W1   = (const float*)d_in[6];
    const float* img_b1   = (const float*)d_in[7];
    const float* img_W2   = (const float*)d_in[8];
    const float* img_b2   = (const float*)d_in[9];
    const float* usr_W1   = (const float*)d_in[10];
    const float* usr_b1   = (const float*)d_in[11];
    const float* usr_W2   = (const float*)d_in[12];
    const float* usr_b2   = (const float*)d_in[13];
    const float* iiW1_img = (const float*)d_in[14];
    const float* iib1_img = (const float*)d_in[15];
    const float* iiW2_img = (const float*)d_in[16];
    const float* iib2_img = (const float*)d_in[17];
    const float* iiW1_usr = (const float*)d_in[18];
    const float* iib1_usr = (const float*)d_in[19];
    const float* iiW2_usr = (const float*)d_in[20];
    const float* iib2_usr = (const float*)d_in[21];
    const float* gWih_img = (const float*)d_in[22];
    const float* gWhh_img = (const float*)d_in[23];
    const float* gbih_img = (const float*)d_in[24];
    const float* gbhh_img = (const float*)d_in[25];
    const float* gWih_usr = (const float*)d_in[26];
    const float* gWhh_usr = (const float*)d_in[27];
    const float* gbih_usr = (const float*)d_in[28];
    const float* gbhh_usr = (const float*)d_in[29];
    const float* corr_W   = (const float*)d_in[30];
    const float* corr_b   = (const float*)d_in[31];
    const float* dist_W1  = (const float*)d_in[32];
    const float* dist_b1  = (const float*)d_in[33];
    const float* dist_W2  = (const float*)d_in[34];
    const float* dist_b2  = (const float*)d_in[35];

    char* ws = (char*)d_ws;
    const size_t NBA = (size_t)B_SZ * NUM_ATTR * DD * 4;  // 6 291 456 B
    float* attr_img      = (float*)(ws + 0 * NBA);
    float* attr_user     = (float*)(ws + 1 * NBA);
    float* internal_img  = (float*)(ws + 2 * NBA);
    float* internal_user = (float*)(ws + 3 * NBA);
    float* fused_img     = (float*)(ws + 4 * NBA);
    float* fused_user    = (float*)(ws + 5 * NBA);
    char* p = ws + 6 * NBA;
    float* img_sum  = (float*)p;  p += (size_t)B_SZ * DD * 4;
    float* user_sum = (float*)p;  p += (size_t)B_SZ * DD * 4;
    float* direct   = (float*)p;  p += (size_t)B_SZ * 4;
    _Float16* W1p_img  = (_Float16*)p; p += 64 * 256 * 2;
    _Float16* W2p_img  = (_Float16*)p; p += 256 * 64 * 2;
    _Float16* W1p_usr  = (_Float16*)p; p += 64 * 256 * 2;
    _Float16* W2p_usr  = (_Float16*)p; p += 256 * 64 * 2;
    _Float16* Wihp_img = (_Float16*)p; p += 64 * 192 * 2;
    _Float16* Whhp_img = (_Float16*)p; p += 64 * 192 * 2;
    _Float16* Wihp_usr = (_Float16*)p; p += 64 * 192 * 2;
    _Float16* Whhp_usr = (_Float16*)p; p += 64 * 192 * 2;

    // 1. pack weights into WMMA B-fragment order (f16)
    pack_wmma_b<<<64, 256, 0, stream>>>(iiW1_img, W1p_img, 64, 256);
    pack_wmma_b<<<64, 256, 0, stream>>>(iiW2_img, W2p_img, 256, 64);
    pack_wmma_b<<<64, 256, 0, stream>>>(iiW1_usr, W1p_usr, 64, 256);
    pack_wmma_b<<<64, 256, 0, stream>>>(iiW2_usr, W2p_usr, 256, 64);
    pack_wmma_b<<<48, 256, 0, stream>>>(gWih_img, Wihp_img, 64, 192);
    pack_wmma_b<<<48, 256, 0, stream>>>(gWhh_img, Whhp_img, 64, 192);
    pack_wmma_b<<<48, 256, 0, stream>>>(gWih_usr, Wihp_usr, 64, 192);
    pack_wmma_b<<<48, 256, 0, stream>>>(gWhh_usr, Whhp_usr, 64, 192);

    // 2. NIMA heads + softmax + direct head + attr_img MLP
    nima_direct_attrimg_kernel<<<B_SZ, 256, 0, stream>>>(
        images, nima_Wl, nima_bl, nima_Wa, nima_ba,
        dist_W1, dist_b1, dist_W2, dist_b2,
        img_W1, img_b1, img_W2, img_b2, attr_img, direct);

    // 3. attr_user MLP
    attr_user_kernel<<<B_SZ, 256, 0, stream>>>(
        traits, usr_W1, usr_b1, usr_W2, usr_b2, attr_user);

    // 4. per-batch attribute sums
    attr_sums_kernel<<<B_SZ, 128, 0, stream>>>(attr_img, attr_user, img_sum, user_sum);

    // 5. internal interactions (WMMA hot spot)
    const uint32_t sh_int = 182784;
    internal_interaction_kernel<<<B_SZ, 256, sh_int, stream>>>(
        attr_img, W1p_img, iib1_img, W2p_img, iib2_img, internal_img);
    internal_interaction_kernel<<<B_SZ, 256, sh_int, stream>>>(
        attr_user, W1p_usr, iib1_usr, W2p_usr, iib2_usr, internal_user);

    // 6. GRU fusion via WMMA (192 blocks x 128 rows)
    const uint32_t sh_gru = 87552;
    const int gru_blocks = (B_SZ * NUM_ATTR) / 128;  // 192
    gru_wmma_kernel<<<gru_blocks, 256, sh_gru, stream>>>(
        attr_img, internal_img, user_sum,
        Wihp_img, Whhp_img, gbih_img, gbhh_img, fused_img);
    gru_wmma_kernel<<<gru_blocks, 256, sh_gru, stream>>>(
        attr_user, internal_user, img_sum,
        Wihp_usr, Whhp_usr, gbih_usr, gbhh_usr, fused_user);

    // 7. final reduce
    final_kernel<<<B_SZ, 64, 0, stream>>>(
        fused_img, fused_user, corr_W, corr_b, direct, (float*)d_out);
}
